// DiSAN_38946763440234
// MI455X (gfx1250) — matrix operand order repaired and checked by reading, not compiled, not run on hardware
//
#include <hip/hip_runtime.h>
#include <hip/hip_bf16.h>

// ---------------------------------------------------------------------------
// DiSAN on MI455X (gfx1250): all matmuls via v_wmma_f32_16x16x32_bf16.
// Double-buffered LDS pipeline, 128x128 tiles for the big GEMMs.
// ---------------------------------------------------------------------------

typedef __attribute__((ext_vector_type(16))) __bf16 v16bf;
typedef __attribute__((ext_vector_type(8)))  float  v8f;

#define BK 32
#define LDSPAD 40   // halves per LDS row (80B: 16B-aligned, staggers banks)

// ---------------------------------------------------------------------------
// Generic bf16 GEMM: C = epi( (A @ B^T + bias) * scale )
//   A: [M, Kdim] bf16 row-major (optionally split at 'ksplit' between A0/A1)
//   B: !bkn -> [N, Kdim] row-major (weights / K-matrix);  bkn -> [Kdim, N]
//      row-major (V-matrix), transposed while staging into LDS.
//   C: f32 (cf32=1) or bf16. Batched via blockIdx.z. epi: 0=id,1=sigmoid,2=tanh
// Tile: TBM x TBN x 32, NTHR threads (= 2*TBM = 2*TBN), wave w owns rows
// [16w,16w+16) x TBN cols. Double-buffered LDS with register prefetch.
// ---------------------------------------------------------------------------
template<int TBM, int TBN, int NTHR>
__global__ __launch_bounds__(NTHR)
void gemm_bf16_kernel(const __bf16* __restrict__ A0, const __bf16* __restrict__ A1,
                      int ksplit, int lda,
                      const __bf16* __restrict__ Bm, int ldb, int bkn,
                      const float* __restrict__ bias,
                      void* __restrict__ Cv, int ldc, int cf32, int epi, float scale,
                      int M, int N, int Kdim,
                      long long strideA, long long strideB, long long strideC)
{
    constexpr int NT = TBN / 16;                 // wmma tiles per wave (cols)
    __shared__ __align__(16) __bf16 lA[2 * TBM * LDSPAD];
    __shared__ __align__(16) __bf16 lB[2 * TBN * LDSPAD];

    const int tid  = threadIdx.x;
    const int wave = tid >> 5;
    const int lane = tid & 31;
    const int m0 = blockIdx.y * TBM;
    const int n0 = blockIdx.x * TBN;
    const long long bz = blockIdx.z;

    const __bf16* Ab0 = A0 + bz * strideA;
    const __bf16* Ab1 = A1 + bz * strideA;
    const __bf16* Bb  = Bm + bz * strideB;

    v8f acc[NT] = {};

    // staging indices: NTHR threads, 16 halves (32B) each per tile
    const int ar  = tid >> 1;                    // 0..TBM-1 (A) / 0..TBN-1 (B nk)
    const int akc = (tid & 1) * 16;              // k-chunk 0 / 16
    const int bkk = tid / (TBN / 16);            // 0..31  (bkn mode: k)
    const int bnc = (tid % (TBN / 16)) * 16;     // n-chunk (bkn mode)

    // fragment lane mapping (ISA 7.12.2)
    const int frow = lane & 15;
    const int fhi  = lane >> 4;

    // ---- staging helpers (register prefetch + LDS store) ----
    auto gloadA = [&](int k0, uint4& u0, uint4& u1) {
        u0 = uint4{0,0,0,0}; u1 = uint4{0,0,0,0};
        const int grow = m0 + ar;
        const int kg   = k0 + akc;
        if (grow < M) {
            const __bf16* ap = (kg < ksplit)
                ? (Ab0 + (long long)grow * lda + kg)
                : (Ab1 + (long long)grow * lda + (kg - ksplit));
            const uint4* gp = (const uint4*)ap;
            u0 = gp[0]; u1 = gp[1];
        }
    };
    auto gloadB = [&](int k0, uint4& u0, uint4& u1) {
        u0 = uint4{0,0,0,0}; u1 = uint4{0,0,0,0};
        if (!bkn) {
            const int gn = n0 + ar;
            if (gn < N) {
                const uint4* gp = (const uint4*)(Bb + (long long)gn * ldb + k0 + akc);
                u0 = gp[0]; u1 = gp[1];
            }
        } else {
            const int gk = k0 + bkk;
            if (gk < Kdim && (n0 + bnc) < N) {
                const uint4* gp = (const uint4*)(Bb + (long long)gk * ldb + n0 + bnc);
                u0 = gp[0]; u1 = gp[1];
            }
        }
    };
    auto storeA = [&](int buf, uint4 u0, uint4 u1) {
        uint4* sp = (uint4*)&lA[buf * TBM * LDSPAD + ar * LDSPAD + akc];
        sp[0] = u0; sp[1] = u1;
    };
    auto storeB = [&](int buf, uint4 u0, uint4 u1) {
        if (!bkn) {
            uint4* sp = (uint4*)&lB[buf * TBN * LDSPAD + ar * LDSPAD + akc];
            sp[0] = u0; sp[1] = u1;
        } else {
            union { uint4 u[2]; __bf16 h[16]; } tr;
            tr.u[0] = u0; tr.u[1] = u1;
            #pragma unroll
            for (int j = 0; j < 16; ++j)
                lB[buf * TBN * LDSPAD + (bnc + j) * LDSPAD + bkk] = tr.h[j];
        }
    };

    // ---- prologue: stage K-slice 0 into buffer 0 ----
    uint4 a0, a1, b0, b1;
    gloadA(0, a0, a1);
    gloadB(0, b0, b1);
    storeA(0, a0, a1);
    storeB(0, b0, b1);
    __syncthreads();

    int buf = 0;
    for (int k0 = 0; k0 < Kdim; k0 += BK) {
        const bool more = (k0 + BK) < Kdim;
        if (more) {                       // prefetch next slice into registers
            gloadA(k0 + BK, a0, a1);
            gloadB(k0 + BK, b0, b1);
        }

        // ---- compute current buffer ----
        // A fragment: lanes 0-15: M=lane, K {0..7,16..23}; lanes 16-31: K {8..15,24..31}
        const __bf16* aFragPtr =
            &lA[buf * TBM * LDSPAD + (wave * 16 + frow) * LDSPAD + fhi * 8];
        union { uint4 u[2]; v16bf v; } af;
        af.u[0] = *(const uint4*)aFragPtr;
        af.u[1] = *(const uint4*)(aFragPtr + 16);

        #pragma unroll
        for (int nt = 0; nt < NT; ++nt) {
            // B fragment: lanes 0-15: N=lane, K {0..7,16..23}; lanes 16-31: K {8..15,24..31}
            union { uint4 u[2]; v16bf v; } bfr;
            const __bf16* bp =
                &lB[buf * TBN * LDSPAD + (nt * 16 + frow) * LDSPAD + fhi * 8];
            bfr.u[0] = *(const uint4*)bp;
            bfr.u[1] = *(const uint4*)(bp + 16);
            acc[nt] = __builtin_amdgcn_wmma_f32_16x16x32_bf16(
                false, af.v, false, bfr.v, (short)0, acc[nt], false, false);
        }

        if (more) {                       // store prefetched slice to other buffer
            storeA(buf ^ 1, a0, a1);
            storeB(buf ^ 1, b0, b1);
        }
        __syncthreads();
        buf ^= 1;
    }

    // ---- epilogue: C VGPR r -> row m0+16w + r + 8*(lane>=16), col = n0+16nt+lane%16
    const int mb = m0 + wave * 16 + fhi * 8;
    #pragma unroll
    for (int nt = 0; nt < NT; ++nt) {
        const int col = n0 + nt * 16 + frow;
        if (col >= N) continue;
        const float bv = bias ? bias[col] : 0.0f;
        #pragma unroll
        for (int r = 0; r < 8; ++r) {
            const int row = mb + r;
            if (row < M) {
                float v = (acc[nt][r] + bv) * scale;
                if (epi == 1)      v = 1.0f / (1.0f + __expf(-v));   // sigmoid
                else if (epi == 2) v = tanhf(v);                      // tanh
                const long long off = bz * strideC + (long long)row * ldc + col;
                if (cf32) ((float*)Cv)[off]  = v;
                else      ((__bf16*)Cv)[off] = (__bf16)v;
            }
        }
    }
}

// ---------------------------------------------------------------------------
// xp = bf16(x + pos_emb[s])           (4 elems / thread)
// ---------------------------------------------------------------------------
__global__ void addpos_kernel(const float* __restrict__ x, const float* __restrict__ pos,
                              __bf16* __restrict__ xp, int S, int D, long long total4)
{
    long long i = (long long)blockIdx.x * blockDim.x + threadIdx.x;
    if (i >= total4) return;
    long long e = i * 4;
    int d = (int)(e % D);
    int s = (int)((e / D) % S);
    float4 xv = *(const float4*)(x + e);
    float4 pv = *(const float4*)(pos + (long long)s * D + d);
    __bf16* o = xp + e;
    o[0] = (__bf16)(xv.x + pv.x); o[1] = (__bf16)(xv.y + pv.y);
    o[2] = (__bf16)(xv.z + pv.z); o[3] = (__bf16)(xv.w + pv.w);
}

__global__ void cvt_bf16_kernel(const float* __restrict__ in, __bf16* __restrict__ out,
                                long long n)
{
    long long i = (long long)blockIdx.x * blockDim.x + threadIdx.x;
    if (i < n) out[i] = (__bf16)in[i];
}

// ---------------------------------------------------------------------------
// Row-wise masked softmax: scores f32 [B*S, S] -> P bf16. One wave per row.
// fw (backward=0): keys <= q valid; bw: keys >= q valid.
// ---------------------------------------------------------------------------
__global__ __launch_bounds__(256)
void softmax_causal_kernel(const float* __restrict__ scores, __bf16* __restrict__ P,
                           int S, int backward)
{
    const int row  = blockIdx.x * 8 + (threadIdx.x >> 5);
    const int lane = threadIdx.x & 31;
    const int q  = row % S;
    const int lo = backward ? q : 0;
    const int hi = backward ? S : (q + 1);
    const float* src = scores + (long long)row * S;
    __bf16*      dst = P      + (long long)row * S;

    float m = -3.0e38f;
    for (int k = lane; k < S; k += 32)
        if (k >= lo && k < hi) m = fmaxf(m, src[k]);
    #pragma unroll
    for (int off = 16; off > 0; off >>= 1)
        m = fmaxf(m, __shfl_xor(m, off, 32));

    float sum = 0.0f;
    for (int k = lane; k < S; k += 32)
        if (k >= lo && k < hi) sum += __expf(src[k] - m);
    #pragma unroll
    for (int off = 16; off > 0; off >>= 1)
        sum += __shfl_xor(sum, off, 32);

    const float inv = 1.0f / sum;
    for (int k = lane; k < S; k += 32) {
        float p = (k >= lo && k < hi) ? __expf(src[k] - m) * inv : 0.0f;
        dst[k] = (__bf16)p;
    }
}

// token_reps = g*fw + (1-g)*bw  (f32 out, straight into d_out token region)
__global__ void combine_kernel(const float* __restrict__ gate,
                               const __bf16* __restrict__ afw,
                               const __bf16* __restrict__ abw,
                               float* __restrict__ tok, long long n)
{
    long long i = (long long)blockIdx.x * blockDim.x + threadIdx.x;
    if (i >= n) return;
    float g = gate[i];
    float f = (float)afw[i];
    float b = (float)abw[i];
    tok[i] = g * f + (1.0f - g) * b;
}

// mean over S -> bf16 [B, H]
__global__ void mean_kernel(const float* __restrict__ tok, __bf16* __restrict__ um,
                            int S, int H)
{
    int h = blockIdx.x * blockDim.x + threadIdx.x;
    int b = blockIdx.y;
    const float* p = tok + (long long)b * S * H + h;
    float s = 0.0f;
    for (int i = 0; i < S; ++i) s += p[(long long)i * H];
    um[b * H + h] = (__bf16)(s * (1.0f / (float)S));
}

// ---------------------------------------------------------------------------
// Host launcher
// ---------------------------------------------------------------------------
extern "C" void kernel_launch(void* const* d_in, const int* in_sizes, int n_in,
                              void* d_out, int out_size, void* d_ws, size_t ws_size,
                              hipStream_t stream)
{
    constexpr int B = 32, S = 448, D = 1024, H = 1024;
    constexpr int BS = B * S;                      // 14336
    constexpr int KSPLIT_NONE = 1 << 30;

    const float* x       = (const float*)d_in[0];
    const float* pos_emb = (const float*)d_in[1];
    const float* Wq_fw = (const float*)d_in[2];  const float* bq_fw = (const float*)d_in[3];
    const float* Wk_fw = (const float*)d_in[4];  const float* bk_fw = (const float*)d_in[5];
    const float* Wv_fw = (const float*)d_in[6];  const float* bv_fw = (const float*)d_in[7];
    const float* Wq_bw = (const float*)d_in[8];  const float* bq_bw = (const float*)d_in[9];
    const float* Wk_bw = (const float*)d_in[10]; const float* bk_bw = (const float*)d_in[11];
    const float* Wv_bw = (const float*)d_in[12]; const float* bv_bw = (const float*)d_in[13];
    const float* Wg    = (const float*)d_in[14]; const float* bg    = (const float*)d_in[15];
    const float* Wu    = (const float*)d_in[16]; const float* bu    = (const float*)d_in[17];

    float* out_utt = (float*)d_out;                 // [B, H]
    float* out_tok = out_utt + (long long)B * H;    // [B, S, H]

    // ---- workspace carve (256B aligned slices); total ~291 MB
    char* w = (char*)d_ws;
    auto carve = [&](size_t bytes) {
        char* p = w; w += (bytes + 255) & ~(size_t)255; return p;
    };
    __bf16* xp   = (__bf16*)carve((size_t)BS * D * 2);
    __bf16* wQf  = (__bf16*)carve((size_t)H * D * 2);
    __bf16* wKf  = (__bf16*)carve((size_t)H * D * 2);
    __bf16* wVf  = (__bf16*)carve((size_t)H * D * 2);
    __bf16* wQb  = (__bf16*)carve((size_t)H * D * 2);
    __bf16* wKb  = (__bf16*)carve((size_t)H * D * 2);
    __bf16* wVb  = (__bf16*)carve((size_t)H * D * 2);
    __bf16* wG   = (__bf16*)carve((size_t)H * 2 * H * 2);
    __bf16* wU   = (__bf16*)carve((size_t)H * H * 2);
    __bf16* Qb   = (__bf16*)carve((size_t)BS * H * 2);   // reused fw then bw
    __bf16* Kb   = (__bf16*)carve((size_t)BS * H * 2);
    __bf16* Vb   = (__bf16*)carve((size_t)BS * H * 2);
    float*  scb  = (float*) carve((size_t)B * S * S * 4); // scores, reused
    __bf16* Pb   = (__bf16*)carve((size_t)B * S * S * 2); // probs, reused
    __bf16* attnF= (__bf16*)carve((size_t)BS * H * 2);
    __bf16* attnB= (__bf16*)carve((size_t)BS * H * 2);
    float*  gate = (float*) carve((size_t)BS * H * 4);
    __bf16* uttm = (__bf16*)carve((size_t)B * H * 2);

    auto cvt = [&](const float* src, __bf16* dst, long long n) {
        cvt_bf16_kernel<<<dim3((unsigned)((n + 255) / 256)), 256, 0, stream>>>(src, dst, n);
    };
    // 128x128 tiles, 256 threads (big GEMMs)
    auto gemmL = [&](const __bf16* A0, const __bf16* A1, int ksplit, int lda,
                     const __bf16* Bm, int ldb, int bkn, const float* bias,
                     void* C, int ldc, int cf32, int epi, float scale,
                     int M, int N, int Kdim, int batches,
                     long long sA, long long sB, long long sC) {
        dim3 grid((unsigned)((N + 127) / 128), (unsigned)((M + 127) / 128), (unsigned)batches);
        gemm_bf16_kernel<128, 128, 256><<<grid, 256, 0, stream>>>(
            A0, A1, ksplit, lda, Bm, ldb, bkn, bias, C, ldc, cf32, epi, scale,
            M, N, Kdim, sA, sB, sC);
    };
    // 64x64 tiles, 128 threads (448-sized / tiny GEMMs)
    auto gemmS = [&](const __bf16* A0, const __bf16* A1, int ksplit, int lda,
                     const __bf16* Bm, int ldb, int bkn, const float* bias,
                     void* C, int ldc, int cf32, int epi, float scale,
                     int M, int N, int Kdim, int batches,
                     long long sA, long long sB, long long sC) {
        dim3 grid((unsigned)((N + 63) / 64), (unsigned)((M + 63) / 64), (unsigned)batches);
        gemm_bf16_kernel<64, 64, 128><<<grid, 128, 0, stream>>>(
            A0, A1, ksplit, lda, Bm, ldb, bkn, bias, C, ldc, cf32, epi, scale,
            M, N, Kdim, sA, sB, sC);
    };

    // 1) weights -> bf16
    cvt(Wq_fw, wQf, (long long)H * D);  cvt(Wk_fw, wKf, (long long)H * D);
    cvt(Wv_fw, wVf, (long long)H * D);  cvt(Wq_bw, wQb, (long long)H * D);
    cvt(Wk_bw, wKb, (long long)H * D);  cvt(Wv_bw, wVb, (long long)H * D);
    cvt(Wg, wG, (long long)H * 2 * H);  cvt(Wu, wU, (long long)H * H);

    // 2) xp = x + pos_emb
    {
        long long total4 = (long long)BS * D / 4;
        addpos_kernel<<<dim3((unsigned)((total4 + 255) / 256)), 256, 0, stream>>>(
            x, pos_emb, xp, S, D, total4);
    }

    const float qscale = 1.0f / 32.0f;   // fold 1/sqrt(H) into Q (incl. bias)

    // ---- forward direction ----
    gemmL(xp, xp, KSPLIT_NONE, D, wQf, D, 0, bq_fw, Qb, H, 0, 0, qscale, BS, H, D, 1, 0, 0, 0);
    gemmL(xp, xp, KSPLIT_NONE, D, wKf, D, 0, bk_fw, Kb, H, 0, 0, 1.0f,  BS, H, D, 1, 0, 0, 0);
    gemmL(xp, xp, KSPLIT_NONE, D, wVf, D, 0, bv_fw, Vb, H, 0, 0, 1.0f,  BS, H, D, 1, 0, 0, 0);
    // scores[b] = Q[b] @ K[b]^T   (f32, batched, 448x448 -> 64-tiles)
    gemmS(Qb, Qb, KSPLIT_NONE, H, Kb, H, 0, nullptr, scb, S, 1, 0, 1.0f,
          S, S, H, B, (long long)S * H, (long long)S * H, (long long)S * S);
    softmax_causal_kernel<<<dim3((unsigned)(B * S / 8)), 256, 0, stream>>>(scb, Pb, S, 0);
    // attn_fw[b] = P[b] @ V[b]    (V is [K,N] layout -> bkn=1; N=1024 -> 128-tiles)
    gemmL(Pb, Pb, KSPLIT_NONE, S, Vb, H, 1, nullptr, attnF, H, 0, 0, 1.0f,
          S, H, S, B, (long long)S * S, (long long)S * H, (long long)S * H);

    // ---- backward direction (reuse Q/K/V/scores/P buffers) ----
    gemmL(xp, xp, KSPLIT_NONE, D, wQb, D, 0, bq_bw, Qb, H, 0, 0, qscale, BS, H, D, 1, 0, 0, 0);
    gemmL(xp, xp, KSPLIT_NONE, D, wKb, D, 0, bk_bw, Kb, H, 0, 0, 1.0f,  BS, H, D, 1, 0, 0, 0);
    gemmL(xp, xp, KSPLIT_NONE, D, wVb, D, 0, bv_bw, Vb, H, 0, 0, 1.0f,  BS, H, D, 1, 0, 0, 0);
    gemmS(Qb, Qb, KSPLIT_NONE, H, Kb, H, 0, nullptr, scb, S, 1, 0, 1.0f,
          S, S, H, B, (long long)S * H, (long long)S * H, (long long)S * S);
    softmax_causal_kernel<<<dim3((unsigned)(B * S / 8)), 256, 0, stream>>>(scb, Pb, S, 1);
    gemmL(Pb, Pb, KSPLIT_NONE, S, Vb, H, 1, nullptr, attnB, H, 0, 0, 1.0f,
          S, H, S, B, (long long)S * S, (long long)S * H, (long long)S * H);

    // ---- gate = sigmoid(concat(attnF, attnB) @ Wg^T + bg)  (split-A, K=2H)
    gemmL(attnF, attnB, H, H, wG, 2 * H, 0, bg, gate, H, 1, 1, 1.0f, BS, H, 2 * H, 1, 0, 0, 0);

    // ---- token_reps = g*fw + (1-g)*bw  -> d_out token region
    {
        long long n = (long long)BS * H;
        combine_kernel<<<dim3((unsigned)((n + 255) / 256)), 256, 0, stream>>>(
            gate, attnF, attnB, out_tok, n);
    }

    // ---- utt_vec = tanh(mean_s(token_reps) @ Wu^T + bu)
    mean_kernel<<<dim3((unsigned)(H / 256), (unsigned)B), 256, 0, stream>>>(out_tok, uttm, S, H);
    gemmS(uttm, uttm, KSPLIT_NONE, H, wU, H, 0, bu, out_utt, H, 1, 2, 1.0f,
          B, H, H, 1, 0, 0, 0);
}
